// AttentionDecoder_4492535792555
// MI455X (gfx1250) — compile-verified, matrix operand order checked by reading, not run on hardware
//
#include <hip/hip_runtime.h>
#include <hip/hip_bf16.h>

// ---------------------------------------------------------------------------
// AttentionDecoder for MI455X (gfx1250): bf16 WMMA GEMMs + fused elementwise.
// B=128, T=256, D=H=V=1024, S=128 sequential steps.
// Working set (h_bf16 + wh_bf16 + bf16 weights ~165MB) fits the 192MB L2;
// per-step GEMMs are K-split across gridDim.z into deterministic f32 partial
// slices to maximize WGP occupancy (M is only 128); K loop fully unrolled so
// loads become immediate-offset clauses pipelined against the WMMAs.
// ---------------------------------------------------------------------------

typedef __attribute__((ext_vector_type(16))) __bf16 v16bf;
typedef __attribute__((ext_vector_type(8)))  float  v8f;

constexpr int B_ = 128, T_ = 256, D_ = 1024, H_ = 1024, V_ = 1024, S_ = 128;

// ------------------------------- helpers -----------------------------------

__global__ void f32_to_bf16_k(const float* __restrict__ in, __bf16* __restrict__ out, int n) {
    int i = blockIdx.x * blockDim.x + threadIdx.x;
    if (i < n) out[i] = (__bf16)in[i];
}

__global__ void add_vec_k(const float* __restrict__ a, const float* __restrict__ b,
                          float* __restrict__ o, int n) {
    int i = blockIdx.x * blockDim.x + threadIdx.x;
    if (i < n) o[i] = a[i] + b[i];
}

// zero y0 slice of x = [c, y] concat buffer (columns 1024..2047)
__global__ void init_y0_k(__bf16* __restrict__ x) {
    int idx = blockIdx.x * blockDim.x + threadIdx.x;   // B*V threads
    int b = idx >> 10, j = idx & 1023;
    x[(size_t)b * 2048 + 1024 + j] = (__bf16)0.0f;
}

// ------------------------- WMMA GEMM (single pass) --------------------------
// C[M,N] = act( A[M,K] @ W[N,K]^T + bias ).  Used for the one-time big GEMMs.
// Block = 256 thr (8 waves); wave w: rows [blk.x*128 + w*16, +16), cols
// [blk.y*64, +64).  Fragment layout per CDNA5 ISA 7.12.2 (16-bit A 16x32):
//   lanes 0-15 : row = lane,    K chunks [0..7] and [16..23]
//   lanes 16-31: row = lane-16, K chunks [8..15] and [24..31]

template<bool BIAS, bool TANH, bool OUTF, bool OUTB>
__global__ void gemm_bf16_wmma(const __bf16* __restrict__ A, int lda,
                               const __bf16* __restrict__ W, int ldw,
                               const float* __restrict__ bias,
                               float* __restrict__ Cf, __bf16* __restrict__ Cb,
                               int ldc, int K)
{
    const int lane  = threadIdx.x & 31;
    const int wave  = threadIdx.x >> 5;
    const int l15   = lane & 15;
    const int koff  = (lane & 16) ? 8 : 0;
    const int mtile = (blockIdx.x * 8 + wave) * 16;
    const int nbase = blockIdx.y * 64;

    v8f acc[4];
    #pragma unroll
    for (int j = 0; j < 4; ++j) acc[j] = (v8f){0,0,0,0,0,0,0,0};

    const __bf16* Arow = A + (size_t)(mtile + l15) * lda;

    for (int k0 = 0; k0 < K; k0 += 32) {
        union { v16bf v; uint4 u[2]; } af;
        af.u[0] = *(const uint4*)(Arow + k0 + koff);
        af.u[1] = *(const uint4*)(Arow + k0 + koff + 16);
        #pragma unroll
        for (int j = 0; j < 4; ++j) {
            const __bf16* Wrow = W + (size_t)(nbase + j * 16 + l15) * ldw;
            union { v16bf v; uint4 u[2]; } bfr;
            bfr.u[0] = *(const uint4*)(Wrow + k0 + koff);
            bfr.u[1] = *(const uint4*)(Wrow + k0 + koff + 16);
            acc[j] = __builtin_amdgcn_wmma_f32_16x16x32_bf16(
                false, af.v, false, bfr.v, (short)0, acc[j], false, false);
        }
    }

    const int mbase = mtile + ((lane & 16) ? 8 : 0);
    #pragma unroll
    for (int j = 0; j < 4; ++j) {
        const int n  = nbase + j * 16 + l15;
        const float bv = BIAS ? bias[n] : 0.0f;
        #pragma unroll
        for (int i = 0; i < 8; ++i) {
            const size_t idx = (size_t)(mbase + i) * ldc + n;
            float v = acc[j][i] + bv;
            if (TANH) v = tanhf(v);
            if (OUTF) Cf[idx] = v;
            if (OUTB) Cb[idx] = (__bf16)v;
        }
    }
}

// ----------------------- WMMA GEMM (K-split partials) -----------------------
// M fixed at 128 (8 waves x 16 rows).  grid = (1, N/64, KS); block z computes
// K range [z*KITERS*32, +(KITERS*32)) and writes its own f32 slice:
//   Cpart[((z*128 + m) * ldc) + n]
// Deterministic (no atomics); consumers sum the slices.  KITERS is a compile-
// time constant so the K loop fully unrolls: all loads become immediate
// offsets off 5 invariant base pointers and pipeline against the WMMAs.

template<int KITERS>
__global__ void gemm_part_k(const __bf16* __restrict__ A, int lda,
                            const __bf16* __restrict__ W, int ldw,
                            float* __restrict__ Cpart, int ldc)
{
    const int lane  = threadIdx.x & 31;
    const int wave  = threadIdx.x >> 5;
    const int l15   = lane & 15;
    const int koff  = (lane & 16) ? 8 : 0;
    const int mtile = wave * 16;
    const int nbase = blockIdx.y * 64;
    const int kbeg  = blockIdx.z * (KITERS * 32);

    v8f acc[4];
    #pragma unroll
    for (int j = 0; j < 4; ++j) acc[j] = (v8f){0,0,0,0,0,0,0,0};

    const __bf16* Arow = A + (size_t)(mtile + l15) * lda + kbeg + koff;
    const __bf16* Wrow[4];
    #pragma unroll
    for (int j = 0; j < 4; ++j)
        Wrow[j] = W + (size_t)(nbase + j * 16 + l15) * ldw + kbeg + koff;

    #pragma unroll
    for (int it = 0; it < KITERS; ++it) {
        union { v16bf v; uint4 u[2]; } af;
        af.u[0] = *(const uint4*)(Arow + it * 32);
        af.u[1] = *(const uint4*)(Arow + it * 32 + 16);
        #pragma unroll
        for (int j = 0; j < 4; ++j) {
            union { v16bf v; uint4 u[2]; } bfr;
            bfr.u[0] = *(const uint4*)(Wrow[j] + it * 32);
            bfr.u[1] = *(const uint4*)(Wrow[j] + it * 32 + 16);
            acc[j] = __builtin_amdgcn_wmma_f32_16x16x32_bf16(
                false, af.v, false, bfr.v, (short)0, acc[j], false, false);
        }
    }

    float* Cs = Cpart + (size_t)blockIdx.z * 128 * ldc;
    const int mbase = mtile + ((lane & 16) ? 8 : 0);
    #pragma unroll
    for (int j = 0; j < 4; ++j) {
        const int n = nbase + j * 16 + l15;
        #pragma unroll
        for (int i = 0; i < 8; ++i)
            Cs[(size_t)(mbase + i) * ldc + n] = acc[j][i];
    }
}

// g[b,j] = sum_{z<4} g_part[z][b,j] + b_s_a[j]
__global__ void reduce_g_k(const float* __restrict__ part, const float* __restrict__ bias,
                           float* __restrict__ out)
{
    const int idx = blockIdx.x * blockDim.x + threadIdx.x;   // B*H threads
    const int j = idx & 1023;
    float v = bias[j];
    #pragma unroll
    for (int z = 0; z < 4; ++z) v += part[(size_t)z * 128 * H_ + idx];
    out[idx] = v;
}

// --------------------------- attention kernels -----------------------------
// scores[b,t] = sum_h tanh(wh[b,t,h] + g[b,h]) * wa[h].  One wave per (b,t)
// row, 8 rows per block (all share the same b since 256 % 8 == 0), g and wa
// staged in LDS, wh loaded as uint4 (8 bf16 per b128 load).
__global__ void attn_scores_k(const __bf16* __restrict__ wh, const float* __restrict__ g,
                              const float* __restrict__ wa, float* __restrict__ scores)
{
    __shared__ float gs[H_];
    __shared__ float was[H_];
    const int tid = threadIdx.x;
    const int b   = (blockIdx.x * 8) >> 8;                   // T = 256
    #pragma unroll
    for (int i = 0; i < 4; ++i) {
        const int j = tid + i * 256;
        gs[j]  = g[(size_t)b * H_ + j];
        was[j] = wa[j];
    }
    __syncthreads();
    const int row  = blockIdx.x * 8 + (tid >> 5);            // row = b*T + t
    const int lane = tid & 31;
    const __bf16* whr = wh + (size_t)row * H_;
    float sum = 0.0f;
    #pragma unroll
    for (int c = 0; c < 4; ++c) {
        const int h0 = c * 256 + lane * 8;
        union { uint4 u; __bf16 h[8]; } w8;
        w8.u = *(const uint4*)(whr + h0);
        #pragma unroll
        for (int i = 0; i < 8; ++i)
            sum += tanhf((float)w8.h[i] + gs[h0 + i]) * was[h0 + i];
    }
    #pragma unroll
    for (int o = 16; o > 0; o >>= 1) sum += __shfl_xor(sum, o, 32);
    if (lane == 0) scores[row] = sum;
}

__global__ void softmax_T_k(const float* __restrict__ scores, float* __restrict__ a)
{
    __shared__ float red[256];
    const int b = blockIdx.x, t = threadIdx.x;               // blockDim = 256 = T
    const float v = scores[b * T_ + t];
    red[t] = v; __syncthreads();
    for (int s = 128; s > 0; s >>= 1) { if (t < s) red[t] = fmaxf(red[t], red[t + s]); __syncthreads(); }
    const float mx = red[0]; __syncthreads();
    const float e = __expf(v - mx);
    red[t] = e; __syncthreads();
    for (int s = 128; s > 0; s >>= 1) { if (t < s) red[t] += red[t + s]; __syncthreads(); }
    a[b * T_ + t] = e / red[0];
}

// c[b,d] = sum_t a[b,t] * h[b,t,d]; thread owns 4 contiguous d (one b64 load
// per t); writes packed bf16 c into both concat buffers.
__global__ void context_k(const float* __restrict__ a, const __bf16* __restrict__ hb,
                          __bf16* __restrict__ x, __bf16* __restrict__ xr)
{
    __shared__ float as[T_];
    const int b = blockIdx.x, tid = threadIdx.x;             // blockDim = 256
    as[tid] = a[b * T_ + tid];
    __syncthreads();
    float acc[4] = {0.f, 0.f, 0.f, 0.f};
    const __bf16* hbase = hb + (size_t)b * T_ * D_ + tid * 4;
    for (int t = 0; t < T_; ++t) {
        const float at = as[t];
        union { uint2 u; __bf16 h[4]; } h4;
        h4.u = *(const uint2*)(hbase + (size_t)t * D_);
        #pragma unroll
        for (int i = 0; i < 4; ++i) acc[i] += at * (float)h4.h[i];
    }
    union { uint2 u; __bf16 h[4]; } cv;
    #pragma unroll
    for (int i = 0; i < 4; ++i) cv.h[i] = (__bf16)acc[i];
    *(uint2*)(x  + (size_t)b * 2048 + tid * 4) = cv.u;
    *(uint2*)(xr + (size_t)b * 2048 + tid * 4) = cv.u;
}

// ------------------------------ GRU gates ----------------------------------
// Sums gi partials (8 slices, K=2048 split) and gh partials (4 slices) + biases.
__global__ void gru_gate_k(const float* __restrict__ gi_part, const float* __restrict__ gh_part,
                           const float* __restrict__ b_ih, const float* __restrict__ b_hh,
                           float* __restrict__ s, __bf16* __restrict__ s_bf,
                           __bf16* __restrict__ xr)
{
    const int idx = blockIdx.x * blockDim.x + threadIdx.x;   // B*H threads
    const int b = idx >> 10, j = idx & 1023;
    float ir = b_ih[j], iz = b_ih[H_ + j], in = b_ih[2 * H_ + j];
    #pragma unroll
    for (int z = 0; z < 8; ++z) {
        const float* p = gi_part + (size_t)(z * 128 + b) * (3 * H_);
        ir += p[j]; iz += p[H_ + j]; in += p[2 * H_ + j];
    }
    float hr = b_hh[j], hz = b_hh[H_ + j], hn = b_hh[2 * H_ + j];
    #pragma unroll
    for (int z = 0; z < 4; ++z) {
        const float* p = gh_part + (size_t)(z * 128 + b) * (3 * H_);
        hr += p[j]; hz += p[H_ + j]; hn += p[2 * H_ + j];
    }
    const float r = 1.0f / (1.0f + __expf(-(ir + hr)));
    const float z = 1.0f / (1.0f + __expf(-(iz + hz)));
    const float n = tanhf(in + r * hn);
    const float sn = (1.0f - z) * n + z * s[idx];
    s[idx]    = sn;
    s_bf[idx] = (__bf16)sn;
    xr[(size_t)b * 2048 + 1024 + j] = (__bf16)sn;
}

// ----------------------- RNN tanh + softmax over V -------------------------
// Sums 12 pre partial slices (8 from xr@Wih, 4 from y@Whh) + combined bias.
__global__ void rnn_softmax_k(const float* __restrict__ pre_part,
                              const float* __restrict__ bsum,
                              float* __restrict__ out, __bf16* __restrict__ x, int step)
{
    __shared__ float red[256];
    const int b = blockIdx.x, tid = threadIdx.x;             // blockDim = 256, 4 elems/thr
    float v[4];
    float mx = -1e30f;
    #pragma unroll
    for (int i = 0; i < 4; ++i) {
        const int j = tid + i * 256;
        float acc = bsum[j];
        #pragma unroll
        for (int z = 0; z < 12; ++z)
            acc += pre_part[(size_t)(z * 128 + b) * V_ + j];
        v[i] = tanhf(acc);
        mx = fmaxf(mx, v[i]);
    }
    red[tid] = mx; __syncthreads();
    for (int s = 128; s > 0; s >>= 1) { if (tid < s) red[tid] = fmaxf(red[tid], red[tid + s]); __syncthreads(); }
    mx = red[0]; __syncthreads();
    float sum = 0.0f;
    #pragma unroll
    for (int i = 0; i < 4; ++i) { v[i] = __expf(v[i] - mx); sum += v[i]; }
    red[tid] = sum; __syncthreads();
    for (int s = 128; s > 0; s >>= 1) { if (tid < s) red[tid] += red[tid + s]; __syncthreads(); }
    const float inv = 1.0f / red[0];
    float* ob = out + ((size_t)b * S_ + step) * V_;
    #pragma unroll
    for (int i = 0; i < 4; ++i) {
        const int j = tid + i * 256;
        const float y = v[i] * inv;
        ob[j] = y;
        x[(size_t)b * 2048 + 1024 + j] = (__bf16)y;          // y_prev for next step
    }
}

// ------------------------------- host side ---------------------------------

extern "C" void kernel_launch(void* const* d_in, const int* in_sizes, int n_in,
                              void* d_out, int out_size, void* d_ws, size_t ws_size,
                              hipStream_t stream)
{
    const float* h        = (const float*)d_in[0];
    const float* w_h_a    = (const float*)d_in[1];
    const float* w_s_a    = (const float*)d_in[2];
    const float* b_s_a    = (const float*)d_in[3];
    const float* w_a      = (const float*)d_in[4];
    const float* w_init_s = (const float*)d_in[5];
    const float* b_init_s = (const float*)d_in[6];
    const float* gru_w_ih = (const float*)d_in[7];
    const float* gru_w_hh = (const float*)d_in[8];
    const float* gru_b_ih = (const float*)d_in[9];
    const float* gru_b_hh = (const float*)d_in[10];
    const float* rnn_w_ih = (const float*)d_in[11];
    const float* rnn_w_hh = (const float*)d_in[12];
    const float* rnn_b_ih = (const float*)d_in[13];
    const float* rnn_b_hh = (const float*)d_in[14];
    float* out = (float*)d_out;

    // ---- workspace bump allocator ----
    char* ws = (char*)d_ws;
    size_t off = 0;
    auto alloc = [&](size_t bytes) -> void* {
        void* p = ws + off;
        off = (off + bytes + 255) & ~(size_t)255;
        return p;
    };
    __bf16* h_bf       = (__bf16*)alloc((size_t)B_ * T_ * D_ * 2);   // 67 MB
    __bf16* wh_bf      = (__bf16*)alloc((size_t)B_ * T_ * H_ * 2);   // 67 MB
    __bf16* w_h_a_bf   = (__bf16*)alloc((size_t)H_ * D_ * 2);
    __bf16* w_s_a_bf   = (__bf16*)alloc((size_t)H_ * H_ * 2);
    __bf16* w_init_bf  = (__bf16*)alloc((size_t)H_ * D_ * 2);
    __bf16* gru_wih_bf = (__bf16*)alloc((size_t)3 * H_ * (V_ + D_) * 2);
    __bf16* gru_whh_bf = (__bf16*)alloc((size_t)3 * H_ * H_ * 2);
    __bf16* rnn_wih_bf = (__bf16*)alloc((size_t)V_ * (H_ + D_) * 2);
    __bf16* rnn_whh_bf = (__bf16*)alloc((size_t)V_ * V_ * 2);
    float*  rnn_b_sum  = (float*) alloc((size_t)V_ * 4);
    float*  s_f32      = (float*) alloc((size_t)B_ * H_ * 4);
    __bf16* s_bf       = (__bf16*)alloc((size_t)B_ * H_ * 2);
    float*  g_part     = (float*) alloc((size_t)4 * B_ * H_ * 4);        // 2 MB
    float*  g_buf      = (float*) alloc((size_t)B_ * H_ * 4);
    float*  scores     = (float*) alloc((size_t)B_ * T_ * 4);
    float*  attn       = (float*) alloc((size_t)B_ * T_ * 4);
    __bf16* x_bf       = (__bf16*)alloc((size_t)B_ * 2048 * 2);      // [c, y]
    __bf16* xr_bf      = (__bf16*)alloc((size_t)B_ * 2048 * 2);      // [c, s_new]
    float*  gi_part    = (float*) alloc((size_t)8 * B_ * 3 * H_ * 4);    // 12.6 MB
    float*  gh_part    = (float*) alloc((size_t)4 * B_ * 3 * H_ * 4);    // 6.3 MB
    float*  pre_part   = (float*) alloc((size_t)12 * B_ * V_ * 4);       // 6.3 MB
    (void)ws_size; (void)n_in; (void)in_sizes; (void)out_size;

    auto cvt = [&](const float* src, __bf16* dst, size_t n) {
        f32_to_bf16_k<<<dim3((unsigned)((n + 255) / 256)), 256, 0, stream>>>(src, dst, (int)n);
    };

    // ---- one-time setup ----
    cvt(h,        h_bf,       (size_t)B_ * T_ * D_);
    cvt(w_h_a,    w_h_a_bf,   (size_t)H_ * D_);
    cvt(w_s_a,    w_s_a_bf,   (size_t)H_ * H_);
    cvt(w_init_s, w_init_bf,  (size_t)H_ * D_);
    cvt(gru_w_ih, gru_wih_bf, (size_t)3 * H_ * (V_ + D_));
    cvt(gru_w_hh, gru_whh_bf, (size_t)3 * H_ * H_);
    cvt(rnn_w_ih, rnn_wih_bf, (size_t)V_ * (H_ + D_));
    cvt(rnn_w_hh, rnn_whh_bf, (size_t)V_ * V_);
    add_vec_k<<<dim3(V_ / 256), 256, 0, stream>>>(rnn_b_ih, rnn_b_hh, rnn_b_sum, V_);
    init_y0_k<<<dim3((B_ * V_) / 256), 256, 0, stream>>>(x_bf);

    // wh = h @ w_h_a^T  (32768 x 1024 x 1024) -> bf16  (4096 blocks, plenty parallel)
    gemm_bf16_wmma<false, false, false, true>
        <<<dim3((B_ * T_) / 128, H_ / 64), 256, 0, stream>>>(
            h_bf, D_, w_h_a_bf, D_, nullptr, nullptr, wh_bf, H_, D_);

    // s0 = tanh(h[:,0,:] @ w_init_s^T + b)   (A rows strided by T*D)
    gemm_bf16_wmma<true, true, true, true>
        <<<dim3(1, H_ / 64), 256, 0, stream>>>(
            h_bf, T_ * D_, w_init_bf, D_, b_init_s, s_f32, s_bf, H_, D_);

    // ---- 128 sequential decode steps ----
    for (int step = 0; step < S_; ++step) {
        // g = s @ w_s_a^T + b_s_a   (K=1024 split 4 ways -> 64 blocks)
        gemm_part_k<8><<<dim3(1, H_ / 64, 4), 256, 0, stream>>>(
            s_bf, H_, w_s_a_bf, H_, g_part, H_);
        reduce_g_k<<<dim3((B_ * H_) / 256), 256, 0, stream>>>(g_part, b_s_a, g_buf);
        // scores / softmax / context
        attn_scores_k<<<dim3((B_ * T_) / 8), 256, 0, stream>>>(wh_bf, g_buf, w_a, scores);
        softmax_T_k<<<dim3(B_), 256, 0, stream>>>(scores, attn);
        context_k<<<dim3(B_), 256, 0, stream>>>(attn, h_bf, x_bf, xr_bf);
        // GRU: gi = [c,y] @ Wih^T (K=2048, 8-way split -> 384 blocks)
        //      gh = s @ Whh^T     (K=1024, 4-way split -> 192 blocks)
        gemm_part_k<8><<<dim3(1, (3 * H_) / 64, 8), 256, 0, stream>>>(
            x_bf, 2048, gru_wih_bf, 2048, gi_part, 3 * H_);
        gemm_part_k<8><<<dim3(1, (3 * H_) / 64, 4), 256, 0, stream>>>(
            s_bf, H_, gru_whh_bf, H_, gh_part, 3 * H_);
        gru_gate_k<<<dim3((B_ * H_) / 256), 256, 0, stream>>>(
            gi_part, gh_part, gru_b_ih, gru_b_hh, s_f32, s_bf, xr_bf);
        // RNN: pre = [c,s_new] @ Wih^T  (slices 0..7)  +  y @ Whh^T (slices 8..11)
        gemm_part_k<8><<<dim3(1, V_ / 64, 8), 256, 0, stream>>>(
            xr_bf, 2048, rnn_wih_bf, 2048, pre_part, V_);
        gemm_part_k<8><<<dim3(1, V_ / 64, 4), 256, 0, stream>>>(
            x_bf + 1024, 2048, rnn_whh_bf, V_, pre_part + (size_t)8 * 128 * V_, V_);
        // y = softmax(tanh(sum partials + bias)) -> d_out[:, step, :] and next y_prev
        rnn_softmax_k<<<dim3(B_), 256, 0, stream>>>(pre_part, rnn_b_sum, out, x_bf, step);
    }
}
